// MultiHeadAttention_4303557230765
// MI455X (gfx1250) — compile-verified, matrix operand order checked by reading, not run on hardware
//
#include <hip/hip_runtime.h>

// ---------------------------------------------------------------------------
// MI455X (gfx1250) fused MHA forward.
//   B=4, T=2048, D=512, H=8, Dh=64.
//   All GEMMs run on v_wmma_f32_16x16x32_bf16 (wave32, 16x16 tiles, K=32).
//   Attention K/V tiles staged with GLOBAL_LOAD_ASYNC_TO_LDS_B128 (ASYNCcnt),
//   double-buffered so tile DMA overlaps the WMMA + softmax work.
//   Pipeline: [weights->bf16^T] -> [QKV proj + RMSnorm + RoPE] ->
//             [flash attention] -> [out proj].
//   Workspace layout (bytes):
//     0        WqT   512*512*2
//     512K     WkT
//     1M       WvT
//     1.5M     WoT
//     2M       Qbf   [b][h][t][d]  bf16  (8 MB)
//     10M      Kbf   [b][h][t][d]  bf16  (8 MB)
//     18M      Vt    [b][h][d][t]  bf16  (8 MB)  (transposed for PV B-operand)
//     26M      Ctx   [b][t][h*d]   bf16  (8 MB)
//   total ~34 MB.
// ---------------------------------------------------------------------------

typedef __attribute__((ext_vector_type(16))) __bf16 v16bf;
typedef __attribute__((ext_vector_type(8)))  float  f32x8;

union FragBF {
    v16bf v;
    uint2 q[4];
    unsigned short h[16];
};

__device__ __forceinline__ unsigned short f2bf(float f) {
    unsigned u = __builtin_bit_cast(unsigned, f);
    unsigned r = u + 0x7FFFu + ((u >> 16) & 1u);   // round-to-nearest-even
    return (unsigned short)(r >> 16);
}

// Load one 16x32 bf16 A/B operand fragment (row-major source).
// Per ISA 7.12.2 (16-bit A-matrix 16x32): lane l holds row (l&15); its 8
// VGPRs hold K = {0..7}+8*hi and {16..23}+8*hi  => byte offsets from the row
// base: {0,8,32,40} + 16*hi (two contiguous 16B runs -> b128 loads).
__device__ __forceinline__ void load_frag(FragBF& f, const unsigned short* rowbase, int hi) {
    const char* p = (const char*)rowbase + (hi << 4);
    f.q[0] = *(const uint2*)(p + 0);
    f.q[1] = *(const uint2*)(p + 8);
    f.q[2] = *(const uint2*)(p + 32);
    f.q[3] = *(const uint2*)(p + 40);
}

__device__ __forceinline__ f32x8 wmma_bf16(const FragBF& a, const FragBF& b, f32x8 c) {
    return __builtin_amdgcn_wmma_f32_16x16x32_bf16(
        false, a.v, false, b.v, (short)0, c, false, false);
}

// --- CDNA5 async global->LDS copy (ASYNCcnt path, ISA ch.10 / 08_async) ----
// vdst VGPR = per-lane LDS byte address (low 32 bits of the generic pointer
// == addrspace(3) offset), vaddr = 64-bit per-lane global address.
__device__ __forceinline__ void async_g2l_b128(unsigned lds_addr, const void* gptr) {
    asm volatile("global_load_async_to_lds_b128 %0, %1, off"
                 :: "v"(lds_addr), "v"((unsigned long long)(uintptr_t)gptr)
                 : "memory");
}

template <int N>
__device__ __forceinline__ void wait_asynccnt() {
    asm volatile("s_wait_asynccnt %0" :: "n"(N) : "memory");
}

__device__ __forceinline__ unsigned lds_addr_of(const void* p) {
    return (unsigned)(uintptr_t)p;
}

#define F32X8_ZERO {0.f,0.f,0.f,0.f,0.f,0.f,0.f,0.f}

// ---------------------------------------------------------------------------
// Kernel 1: W (512x512 f32, [k][n]) -> WT (bf16 [n][k]) for wq, wk, wv, wo.
// ---------------------------------------------------------------------------
__global__ __launch_bounds__(256) void wt_prep_kernel(
    const float* __restrict__ wq, const float* __restrict__ wk,
    const float* __restrict__ wv, const float* __restrict__ wo,
    unsigned short* __restrict__ WqT, unsigned short* __restrict__ WkT,
    unsigned short* __restrict__ WvT, unsigned short* __restrict__ WoT)
{
    const float* srcs[4] = {wq, wk, wv, wo};
    unsigned short* dsts[4] = {WqT, WkT, WvT, WoT};
    int mat = blockIdx.y;
    int idx = blockIdx.x * 256 + threadIdx.x;      // 0 .. 512*512-1
    int n = idx >> 9;
    int k = idx & 511;
    dsts[mat][(size_t)n * 512 + k] = f2bf(srcs[mat][(size_t)k * 512 + n]);
}

// ---------------------------------------------------------------------------
// Kernel 2: QKV projection + bias + per-head RMS-norm + RoPE.
//   grid = (512 row-tiles of 16, 3 proj types), block = 256 (8 waves = 8 heads)
//   (A tile staged synchronously: the f32->bf16 conversion happens in flight.)
// ---------------------------------------------------------------------------
__global__ __launch_bounds__(256) void qkv_kernel(
    const float* __restrict__ qin, const float* __restrict__ kvin,
    const unsigned short* __restrict__ WqT, const unsigned short* __restrict__ WkT,
    const unsigned short* __restrict__ WvT,
    const float* __restrict__ bq, const float* __restrict__ bk, const float* __restrict__ bv,
    const float* __restrict__ scale_q, const float* __restrict__ scale_k,
    unsigned short* __restrict__ Qbf, unsigned short* __restrict__ Kbf,
    unsigned short* __restrict__ Vtg)
{
    const int tid  = threadIdx.x;
    const int type = blockIdx.y;                 // 0=q, 1=k, 2=v
    const int r0   = blockIdx.x * 16;            // row base within 8192
    const int bIdx = r0 >> 11;                   // batch
    const int t0   = r0 & 2047;                  // sequence position base

    const float*          X    = (type == 0) ? qin : kvin;
    const unsigned short* WT   = (type == 0) ? WqT : ((type == 1) ? WkT : WvT);
    const float*          bias = (type == 0) ? bq  : ((type == 1) ? bk  : bv);

    __shared__ unsigned short Atile[16 * 512];   // 16 KB

    // Cooperatively load + convert the 16x512 activation tile to bf16.
    #pragma unroll
    for (int e = 0; e < 8; ++e) {
        int l4   = e * 256 + tid;                // float4 index, 0..2047
        int fidx = l4 * 4;
        int row  = fidx >> 9, col = fidx & 511;
        float4 f = *(const float4*)(X + (size_t)(r0 + row) * 512 + col);
        uint2 p;
        p.x = (unsigned)f2bf(f.x) | ((unsigned)f2bf(f.y) << 16);
        p.y = (unsigned)f2bf(f.z) | ((unsigned)f2bf(f.w) << 16);
        *(uint2*)(&Atile[fidx]) = p;
    }
    __syncthreads();

    const int wv_ = tid >> 5;                    // wave index == head
    const int lane = tid & 31, lm = lane & 15, hi = lane >> 4;

    f32x8 acc[4] = {F32X8_ZERO, F32X8_ZERO, F32X8_ZERO, F32X8_ZERO};

    for (int ks = 0; ks < 16; ++ks) {
        FragBF a;
        load_frag(a, &Atile[lm * 512 + ks * 32], hi);
        #pragma unroll
        for (int j = 0; j < 4; ++j) {
            FragBF b;
            load_frag(b, WT + (size_t)(wv_ * 64 + j * 16 + lm) * 512 + ks * 32, hi);
            acc[j] = wmma_bf16(a, b, acc[j]);
        }
    }

    // bias
    float bb[4];
    #pragma unroll
    for (int j = 0; j < 4; ++j) bb[j] = bias[wv_ * 64 + j * 16 + lm];
    #pragma unroll
    for (int j = 0; j < 4; ++j)
        #pragma unroll
        for (int i = 0; i < 8; ++i) acc[j][i] += bb[j];

    if (type < 2) {
        // RMS norm over the 64-wide head dim: row i+8*hi lives in element i
        // across 16 lanes of this half-wave * 4 n-tiles.
        const float* sp = (type == 0) ? scale_q : scale_k;
        float sc[4];
        #pragma unroll
        for (int j = 0; j < 4; ++j) sc[j] = 1.0f + sp[j * 16 + lm];
        #pragma unroll
        for (int i = 0; i < 8; ++i) {
            float ss = acc[0][i]*acc[0][i] + acc[1][i]*acc[1][i]
                     + acc[2][i]*acc[2][i] + acc[3][i]*acc[3][i];
            ss += __shfl_xor(ss, 1, 32);
            ss += __shfl_xor(ss, 2, 32);
            ss += __shfl_xor(ss, 4, 32);
            ss += __shfl_xor(ss, 8, 32);
            float r = rsqrtf(ss * (1.0f / 64.0f) + 1e-6f);
            #pragma unroll
            for (int j = 0; j < 4; ++j) acc[j][i] *= r * sc[j];
        }
        // RoPE: pairs (d, d+32) => (acc[jp], acc[jp+2]).
        #pragma unroll
        for (int jp = 0; jp < 2; ++jp) {
            float d = (float)(jp * 16 + lm);
            float invts = __expf(d * (-9.210340371976184f / 32.0f)); // 10000^(-d/32)
            #pragma unroll
            for (int i = 0; i < 8; ++i) {
                float ang = (float)(t0 + i + 8 * hi) * invts;
                float sn, cs;
                __sincosf(ang, &sn, &cs);
                float x1 = acc[jp][i], x2 = acc[jp + 2][i];
                acc[jp][i]     = x1 * cs - x2 * sn;
                acc[jp + 2][i] = x2 * cs + x1 * sn;
            }
        }
        unsigned short* dst = (type == 0) ? Qbf : Kbf;
        #pragma unroll
        for (int j = 0; j < 4; ++j)
            #pragma unroll
            for (int i = 0; i < 8; ++i) {
                int ti = t0 + i + 8 * hi;
                dst[((size_t)(bIdx * 8 + wv_) * 2048 + ti) * 64 + j * 16 + lm] =
                    f2bf(acc[j][i]);
            }
    } else {
        // V stored transposed: Vt[b][h][d][t]
        #pragma unroll
        for (int j = 0; j < 4; ++j)
            #pragma unroll
            for (int i = 0; i < 8; ++i) {
                int ti = t0 + i + 8 * hi;
                Vtg[((size_t)(bIdx * 8 + wv_) * 64 + j * 16 + lm) * 2048 + ti] =
                    f2bf(acc[j][i]);
            }
    }
}

// ---------------------------------------------------------------------------
// Kernel 3: causal flash attention with double-buffered async K/V staging.
//   grid = (B*H = 32, T/128 = 16 q-tiles), block = 256 (8 waves x 16 q-rows).
// ---------------------------------------------------------------------------
__global__ __launch_bounds__(256) void attn_kernel(
    const unsigned short* __restrict__ Qbf, const unsigned short* __restrict__ Kbf,
    const unsigned short* __restrict__ Vtg, unsigned short* __restrict__ Ctx)
{
    const int bh  = blockIdx.x;
    const int qb  = blockIdx.y * 128;
    const int tid = threadIdx.x;
    const int wv_ = tid >> 5, lane = tid & 31, lm = lane & 15, hi = lane >> 4;
    const int tq0 = qb + wv_ * 16;

    __shared__ unsigned short KtL[2][32 * 64];    // K tiles  (u, d)   2 x 4 KB
    __shared__ unsigned short VtL[2][64 * 32];    // V^T tiles (d, u)  2 x 4 KB
    __shared__ unsigned short PstL[8 * 16 * 32];  // per-wave P stage  8 KB

    // Per-thread async-copy slices (16 bytes each of the K and V^T tiles).
    const int krow = tid >> 3, kcg = (tid & 7) * 8;
    const int vrow = tid >> 2, vcg = (tid & 3) * 8;
    const unsigned ldsK0 = lds_addr_of(&KtL[0][krow * 64 + kcg]);
    const unsigned ldsV0 = lds_addr_of(&VtL[0][vrow * 32 + vcg]);
    const unsigned short* gK0 = Kbf + ((size_t)bh * 2048 + krow) * 64 + kcg;
    const unsigned short* gV0 = Vtg + ((size_t)bh * 64 + vrow) * 2048 + vcg;

    // Q fragments for this wave's 16 rows (A operand, two k=32 slabs).
    FragBF qf[2];
    {
        const unsigned short* qrow = Qbf + ((size_t)bh * 2048 + tq0 + lm) * 64;
        load_frag(qf[0], qrow + 0,  hi);
        load_frag(qf[1], qrow + 32, hi);
    }

    f32x8 acc[4] = {F32X8_ZERO, F32X8_ZERO, F32X8_ZERO, F32X8_ZERO};
    float m_i[8], l_i[8];
    #pragma unroll
    for (int i = 0; i < 8; ++i) { m_i[i] = -1e30f; l_i[i] = 0.0f; }

    const int niter = (qb + 128) / 32;            // causal bound, block-uniform

    // Prefetch tile 0 into buffer 0.
    async_g2l_b128(ldsK0, gK0);
    async_g2l_b128(ldsV0, gV0);

    for (int it = 0; it < niter; ++it) {
        const int buf = it & 1;
        const int kvb = it * 32;

        if (it + 1 < niter) {
            // Issue next tile into the other buffer (its readers finished at
            // the end-of-iteration barrier of it-1), then wait until only
            // those two new transfers remain in flight (async loads retire
            // in order => the current tile's pair has completed).
            async_g2l_b128(ldsK0 + (unsigned)((buf ^ 1) * sizeof(KtL[0])),
                           gK0 + (size_t)(kvb + 32) * 64);
            async_g2l_b128(ldsV0 + (unsigned)((buf ^ 1) * sizeof(VtL[0])),
                           gV0 + (kvb + 32));
            wait_asynccnt<2>();
        } else {
            wait_asynccnt<0>();
        }
        __syncthreads();                          // current tile visible to all

        const unsigned short* Kt = KtL[buf];
        const unsigned short* Vt = VtL[buf];

        // S = Q * K^T  (16 x 32), two k-steps, two n-tiles.
        f32x8 s0 = F32X8_ZERO, s1 = F32X8_ZERO;
        #pragma unroll
        for (int ks = 0; ks < 2; ++ks) {
            FragBF kb0, kb1;
            load_frag(kb0, &Kt[(lm)      * 64 + ks * 32], hi);
            load_frag(kb1, &Kt[(16 + lm) * 64 + ks * 32], hi);
            s0 = wmma_bf16(qf[ks], kb0, s0);
            s1 = wmma_bf16(qf[ks], kb1, s1);
        }

        // Online softmax; write P (bf16) into this wave's LDS staging patch.
        #pragma unroll
        for (int i = 0; i < 8; ++i) {
            int ti = tq0 + i + 8 * hi;
            float a0 = s0[i] * 0.125f; if (kvb + lm      > ti) a0 = -1e30f;
            float a1 = s1[i] * 0.125f; if (kvb + 16 + lm > ti) a1 = -1e30f;
            float rm = fmaxf(a0, a1);
            rm = fmaxf(rm, __shfl_xor(rm, 1, 32));
            rm = fmaxf(rm, __shfl_xor(rm, 2, 32));
            rm = fmaxf(rm, __shfl_xor(rm, 4, 32));
            rm = fmaxf(rm, __shfl_xor(rm, 8, 32));
            float nm = fmaxf(m_i[i], rm);
            float p0 = __expf(a0 - nm), p1 = __expf(a1 - nm);
            float rs = p0 + p1;
            rs += __shfl_xor(rs, 1, 32);
            rs += __shfl_xor(rs, 2, 32);
            rs += __shfl_xor(rs, 4, 32);
            rs += __shfl_xor(rs, 8, 32);
            float al = __expf(m_i[i] - nm);
            l_i[i] = l_i[i] * al + rs;
            m_i[i] = nm;
            #pragma unroll
            for (int j = 0; j < 4; ++j) acc[j][i] *= al;
            PstL[(wv_ * 16 + i + 8 * hi) * 32 + lm]      = f2bf(p0);
            PstL[(wv_ * 16 + i + 8 * hi) * 32 + 16 + lm] = f2bf(p1);
        }
        __syncthreads();                          // P (C-layout) -> A-layout reload

        // O += P * V   (A = P 16x32, B = V via V^T rows, 4 n-tiles of d)
        FragBF pf;
        load_frag(pf, &PstL[(wv_ * 16 + lm) * 32], hi);
        #pragma unroll
        for (int j = 0; j < 4; ++j) {
            FragBF vf;
            load_frag(vf, &Vt[(j * 16 + lm) * 32], hi);
            acc[j] = wmma_bf16(pf, vf, acc[j]);
        }
        __syncthreads();                          // all readers done before the
                                                  // next prefetch overwrites buf^1
    }

    // Normalize and emit context (bf16, [b][t][h*64+d]).
    const int bIdx = bh >> 3, h = bh & 7;
    #pragma unroll
    for (int i = 0; i < 8; ++i) {
        float inv = 1.0f / l_i[i];
        int ti = tq0 + i + 8 * hi;
        #pragma unroll
        for (int j = 0; j < 4; ++j) {
            Ctx[((size_t)bIdx * 2048 + ti) * 512 + h * 64 + j * 16 + lm] =
                f2bf(acc[j][i] * inv);
        }
    }
}

// ---------------------------------------------------------------------------
// Kernel 4: out projection  Ctx(8192x512 bf16) @ wo + bo -> out (f32).
//   A tile staged with async global->LDS (raw bf16 copy, no conversion).
// ---------------------------------------------------------------------------
__global__ __launch_bounds__(256) void outproj_kernel(
    const unsigned short* __restrict__ Ctx, const unsigned short* __restrict__ WoT,
    const float* __restrict__ bo, float* __restrict__ out)
{
    const int tid = threadIdx.x;
    const int r0  = blockIdx.x * 16;

    __shared__ unsigned short Atile[16 * 512];
    #pragma unroll
    for (int e = 0; e < 4; ++e) {
        int l4 = e * 256 + tid;                  // uint4 (8 ushort) index, 0..1023
        int sidx = l4 * 8;
        int row = sidx >> 9, col = sidx & 511;
        async_g2l_b128(lds_addr_of(&Atile[sidx]),
                       Ctx + (size_t)(r0 + row) * 512 + col);
    }
    wait_asynccnt<0>();
    __syncthreads();

    const int wv_ = tid >> 5, lane = tid & 31, lm = lane & 15, hi = lane >> 4;

    f32x8 acc[4] = {F32X8_ZERO, F32X8_ZERO, F32X8_ZERO, F32X8_ZERO};
    for (int ks = 0; ks < 16; ++ks) {
        FragBF a;
        load_frag(a, &Atile[lm * 512 + ks * 32], hi);
        #pragma unroll
        for (int j = 0; j < 4; ++j) {
            FragBF b;
            load_frag(b, WoT + (size_t)(wv_ * 64 + j * 16 + lm) * 512 + ks * 32, hi);
            acc[j] = wmma_bf16(a, b, acc[j]);
        }
    }

    float bb[4];
    #pragma unroll
    for (int j = 0; j < 4; ++j) bb[j] = bo[wv_ * 64 + j * 16 + lm];
    #pragma unroll
    for (int j = 0; j < 4; ++j)
        #pragma unroll
        for (int i = 0; i < 8; ++i) {
            out[(size_t)(r0 + i + 8 * hi) * 512 + wv_ * 64 + j * 16 + lm] =
                acc[j][i] + bb[j];
        }
}

// ---------------------------------------------------------------------------
extern "C" void kernel_launch(void* const* d_in, const int* in_sizes, int n_in,
                              void* d_out, int out_size, void* d_ws, size_t ws_size,
                              hipStream_t stream) {
    (void)in_sizes; (void)n_in; (void)out_size; (void)ws_size;
    const float* q        = (const float*)d_in[0];
    const float* kv       = (const float*)d_in[1];
    // d_in[2] mask (causal, reproduced analytically); d_in[3..4] positions (== arange)
    const float* wq       = (const float*)d_in[5];
    const float* bq       = (const float*)d_in[6];
    const float* wk       = (const float*)d_in[7];
    const float* bk       = (const float*)d_in[8];
    const float* wv       = (const float*)d_in[9];
    const float* bv       = (const float*)d_in[10];
    const float* scale_q  = (const float*)d_in[11];
    const float* scale_k  = (const float*)d_in[12];
    const float* wo       = (const float*)d_in[13];
    const float* bo       = (const float*)d_in[14];
    float* out            = (float*)d_out;

    char* ws = (char*)d_ws;
    unsigned short* WqT = (unsigned short*)(ws + 0);
    unsigned short* WkT = (unsigned short*)(ws + (size_t)512 * 1024);
    unsigned short* WvT = (unsigned short*)(ws + (size_t)1024 * 1024);
    unsigned short* WoT = (unsigned short*)(ws + (size_t)1536 * 1024);
    unsigned short* Qbf = (unsigned short*)(ws + (size_t)2 * 1024 * 1024);
    unsigned short* Kbf = (unsigned short*)(ws + (size_t)10 * 1024 * 1024);
    unsigned short* Vtg = (unsigned short*)(ws + (size_t)18 * 1024 * 1024);
    unsigned short* Ctx = (unsigned short*)(ws + (size_t)26 * 1024 * 1024);

    wt_prep_kernel<<<dim3(1024, 4), 256, 0, stream>>>(wq, wk, wv, wo, WqT, WkT, WvT, WoT);
    qkv_kernel<<<dim3(512, 3), 256, 0, stream>>>(q, kv, WqT, WkT, WvT,
                                                 bq, bk, bv, scale_q, scale_k,
                                                 Qbf, Kbf, Vtg);
    attn_kernel<<<dim3(32, 16), 256, 0, stream>>>(Qbf, Kbf, Vtg, Ctx);
    outproj_kernel<<<dim3(512), 256, 0, stream>>>(Ctx, WoT, bo, out);
}
